// Gnn_81114752352453
// MI455X (gfx1250) — matrix operand order repaired and checked by reading, not compiled, not run on hardware
//
#include <hip/hip_runtime.h>

typedef __bf16 bf16;
typedef __attribute__((ext_vector_type(16))) __bf16 v16bf;
typedef __attribute__((ext_vector_type(8)))  float  v8f;

#define N  512
#define D  256
#define R  4
#define FF 512
#define DH 128

union Frag {
    v16bf v;
    bf16  h[16];
    uint4 q[2];
};

// ---------------------------------------------------------------------------
// W0cT[f][k] = W_sc0[2D + k][f]  (bf16), so WMMA B-fragments are K-contiguous.
// ---------------------------------------------------------------------------
__global__ __launch_bounds__(256) void k_w0cT(const float* __restrict__ W0,
                                              bf16* __restrict__ W0cT) {
    int idx = blockIdx.x * 256 + threadIdx.x;   // FF*D
    int k = idx & (D - 1);
    int f = idx >> 8;
    W0cT[f * D + k] = (bf16)W0[(2 * D + k) * FF + f];
}

// ---------------------------------------------------------------------------
// Pa[j,f] = cand[j]·W_sc0[0:D,f] ; Pb[j,f] = cand[j]·W_sc0[D:2D,f]
// ---------------------------------------------------------------------------
__global__ __launch_bounds__(256) void k_pab(const float* __restrict__ cand,
                                             const float* __restrict__ W0,
                                             float* __restrict__ Pa,
                                             float* __restrict__ Pb) {
    int idx = blockIdx.x * 256 + threadIdx.x;   // 2*N*FF
    int f = idx & (FF - 1);
    int j = (idx >> 9) & (N - 1);
    int which = idx >> 18;
    const float* w = W0 + (size_t)which * D * FF;
    const float* c = cand + j * D;
    float s = 0.f;
    for (int k = 0; k < D; ++k) s += c[k] * w[k * FF + f];
    (which ? Pb : Pa)[j * FF + f] = s;
}

// ---------------------------------------------------------------------------
// Pair scorer: per block = one i, 128 j's (8 waves x 16 rows).
// h[j,f] = relu( wmma{ (cand_j ⊙ cand_i) @ W0c } + Pa[j,f] + Pb[i,f] + b0[f] )
// scores[j,r] = h·W1 + b1 -> softmax over R -> A[r,i,j] (diag zeroed).
// B-tiles staged via double-buffered GLOBAL_LOAD_ASYNC_TO_LDS_B128 (ASYNCcnt).
// ---------------------------------------------------------------------------
__global__ __launch_bounds__(256) void k_scorer(const float* __restrict__ cand,
                                                const float* __restrict__ Pa,
                                                const float* __restrict__ Pb,
                                                const float* __restrict__ b0,
                                                const float* __restrict__ W1,
                                                const float* __restrict__ b1,
                                                const bf16*  __restrict__ W0cT,
                                                float* __restrict__ A) {
    const int i    = blockIdx.y;
    const int jb   = blockIdx.x;          // 0..3
    const int tid  = threadIdx.x;
    const int wave = tid >> 5;
    const int lane = tid & 31;
    const int m    = lane & 15;
    const int hi   = lane >> 4;           // upper half-wave flag

    __shared__ float s_ci[D];             // cand[i,:]
    __shared__ float s_pbb[FF];           // Pb[i,f] + b0[f]
    __shared__ float s_w1[FF * R];        // W_sc1
    __shared__ bf16  s_b[2][16 * D];      // double-buffered B-tile (2 x 8KB)

    // --- async B-tile stage: 8KB per tile = 512 x b128, 2 per thread ---
    const uint32_t lds_b   = (uint32_t)(uintptr_t)(&s_b[0][0]);
    const uint64_t gbase   = (uint64_t)(uintptr_t)W0cT;
    const uint32_t tid16   = (uint32_t)tid * 16u;
    auto issue_tile = [&](int ft, int buf) {
        uint32_t l0 = lds_b + (uint32_t)buf * (16u * D * 2u) + tid16;
        uint32_t v0 = (uint32_t)ft * (D * 2u) + tid16;   // byte offset into W0cT
        asm volatile("global_load_async_to_lds_b128 %0, %1, %2"
                     :: "v"(l0), "v"(v0), "s"(gbase) : "memory");
        asm volatile("global_load_async_to_lds_b128 %0, %1, %2"
                     :: "v"(l0 + 4096u), "v"(v0 + 4096u), "s"(gbase) : "memory");
    };

    issue_tile(0, 0);   // prefetch tile 0; overlaps with setup below

    for (int k = tid; k < D; k += 256)       s_ci[k]  = cand[i * D + k];
    for (int f = tid; f < FF; f += 256)      s_pbb[f] = Pb[i * FF + f] + b0[f];
    for (int t = tid; t < FF * R; t += 256)  s_w1[t]  = W1[t];
    __syncthreads();

    const int j0   = jb * 128 + wave * 16;
    const int jrow = j0 + m;
    const float* crow = cand + jrow * D;

    // Build A-fragments (bf16, ISA 16-bit A layout) in registers, reused over all f-tiles.
    Frag a[8];
#pragma unroll
    for (int kt = 0; kt < 8; ++kt) {
        int kb = kt * 32 + hi * 8;
#pragma unroll
        for (int half = 0; half < 2; ++half) {
            int k = kb + half * 16;
            const float* cj = crow + k;
#pragma unroll
            for (int e = 0; e < 8; ++e)
                a[kt].h[half * 8 + e] = (bf16)(cj[e] * s_ci[k + e]);
        }
    }

    float acc[8][R];
#pragma unroll
    for (int v = 0; v < 8; ++v)
#pragma unroll
        for (int r = 0; r < R; ++r) acc[v][r] = 0.f;

    const int T = FF / 16;   // 32 f-tiles
    for (int t = 0; t < T; ++t) {
        __syncthreads();     // everyone done reading buf[(t+1)&1] (iter t-1 compute)
        if (t + 1 < T) {
            issue_tile((t + 1) * 16, (t + 1) & 1);
            asm volatile("s_wait_asynccnt 0x2" ::: "memory");   // tile t complete
        } else {
            asm volatile("s_wait_asynccnt 0x0" ::: "memory");
        }
        __syncthreads();     // tile t visible to all waves

        const int ft = t * 16;
        v8f c = {0.f, 0.f, 0.f, 0.f, 0.f, 0.f, 0.f, 0.f};
        const bf16* brow = &s_b[t & 1][0] + m * D + hi * 8;
#pragma unroll
        for (int kt = 0; kt < 8; ++kt) {
            Frag b;
            b.q[0] = *(const uint4*)(brow + kt * 32);
            b.q[1] = *(const uint4*)(brow + kt * 32 + 16);
            c = __builtin_amdgcn_wmma_f32_16x16x32_bf16(
                    false, a[kt].v, false, b.v, (short)0, c, false, false);
        }

        // fused epilogue: bias + relu + tiny FF->R projection
        int f = ft + m;
        float pbbf = s_pbb[f];
        const float* w1f = s_w1 + f * R;
#pragma unroll
        for (int v = 0; v < 8; ++v) {
            int j = j0 + hi * 8 + v;
            float hval = c[v] + Pa[j * FF + f] + pbbf;
            hval = fmaxf(hval, 0.f);
#pragma unroll
            for (int r = 0; r < R; ++r) acc[v][r] += hval * w1f[r];
        }
    }

    // reduce partial scores across the 16 f-lanes of each half-wave
#pragma unroll
    for (int v = 0; v < 8; ++v)
#pragma unroll
        for (int r = 0; r < R; ++r) {
            float s = acc[v][r];
            s += __shfl_xor(s, 1, 32);
            s += __shfl_xor(s, 2, 32);
            s += __shfl_xor(s, 4, 32);
            s += __shfl_xor(s, 8, 32);
            acc[v][r] = s;
        }

    if (m == 0) {   // lanes 0 (rows j0..j0+7) and 16 (rows j0+8..j0+15)
        float bb0 = b1[0], bb1 = b1[1], bb2 = b1[2], bb3 = b1[3];
#pragma unroll
        for (int v = 0; v < 8; ++v) {
            int j = j0 + hi * 8 + v;
            float s0 = acc[v][0] + bb0, s1 = acc[v][1] + bb1;
            float s2 = acc[v][2] + bb2, s3 = acc[v][3] + bb3;
            float mx = fmaxf(fmaxf(s0, s1), fmaxf(s2, s3));
            float e0 = __expf(s0 - mx), e1 = __expf(s1 - mx);
            float e2 = __expf(s2 - mx), e3 = __expf(s3 - mx);
            float inv = 1.f / (e0 + e1 + e2 + e3);
            float p0 = e0 * inv, p1 = e1 * inv, p2 = e2 * inv, p3 = e3 * inv;
            if (j == i) { p0 = p1 = p2 = p3 = 0.f; }
            A[(size_t)0 * N * N + i * N + j] = p0;
            A[(size_t)1 * N * N + i * N + j] = p1;
            A[(size_t)2 * N * N + i * N + j] = p2;
            A[(size_t)3 * N * N + i * N + j] = p3;
        }
    }
}

// ---------------------------------------------------------------------------
// sup[dir][r][n][e] = x[n,:] · W_{fw|bw}[r][:,e]
// ---------------------------------------------------------------------------
__global__ __launch_bounds__(256) void k_sup(const float* __restrict__ x,
                                             const float* __restrict__ Wfw,
                                             const float* __restrict__ Wbw,
                                             float* __restrict__ sup) {
    int idx = blockIdx.x * 256 + threadIdx.x;   // 2*R*N*DH
    int e = idx & (DH - 1);
    int n = (idx >> 7) & (N - 1);
    int r = (idx >> 16) & (R - 1);
    int dir = idx >> 18;
    const float* W  = (dir ? Wbw : Wfw) + (size_t)r * D * DH;
    const float* xr = x + n * D;
    float s = 0.f;
    for (int k = 0; k < D; ++k) s += xr[k] * W[k * DH + e];
    sup[idx] = s;
}

// ---------------------------------------------------------------------------
// y[i,d] = relu( sum_r (A^T@sup_bw | A@sup_fw)[r,i,d] + biases )
// ---------------------------------------------------------------------------
__global__ __launch_bounds__(256) void k_gcn(const float* __restrict__ A,
                                             const float* __restrict__ sup,
                                             const float* __restrict__ b_fw,
                                             const float* __restrict__ b_bw,
                                             float* __restrict__ y) {
    int idx = blockIdx.x * 256 + threadIdx.x;   // N*D
    int d = idx & (D - 1);
    int i = idx >> 8;
    float s = 0.f;
    if (d < DH) {                                // bw half: AT[r] @ sup_bw
        int e = d;
        const float* supb = sup + (size_t)R * N * DH;
        for (int r = 0; r < R; ++r) {
            s += b_bw[r * DH + e];
            const float* Ar = A + (size_t)r * N * N;
            const float* sb = supb + (size_t)r * N * DH;
            for (int j = 0; j < N; ++j) s += Ar[j * N + i] * sb[j * DH + e];
        }
    } else {                                     // fw half: A[r] @ sup_fw
        int e = d - DH;
        for (int r = 0; r < R; ++r) {
            s += b_fw[r * DH + e];
            const float* Ar = A + (size_t)r * N * N + (size_t)i * N;
            const float* sf = sup + (size_t)r * N * DH;
            for (int j = 0; j < N; ++j) s += Ar[j] * sf[j * DH + e];
        }
    }
    y[idx] = fmaxf(s, 0.f);
}

// ---------------------------------------------------------------------------
// xout[i,d] = y[i,:]·W_lin[:,d] + b_lin[d] + xin[i,d]
// ---------------------------------------------------------------------------
__global__ __launch_bounds__(256) void k_lin(const float* __restrict__ y,
                                             const float* __restrict__ Wl,
                                             const float* __restrict__ bl,
                                             const float* __restrict__ xin,
                                             float* __restrict__ xout) {
    int idx = blockIdx.x * 256 + threadIdx.x;   // N*D
    int d = idx & (D - 1);
    int i = idx >> 8;
    const float* yr = y + i * D;
    float s = bl[d] + xin[idx];
    for (int e = 0; e < D; ++e) s += yr[e] * Wl[e * D + d];
    xout[idx] = s;
}

extern "C" void kernel_launch(void* const* d_in, const int* in_sizes, int n_in,
                              void* d_out, int out_size, void* d_ws, size_t ws_size,
                              hipStream_t stream) {
    (void)in_sizes; (void)n_in; (void)out_size; (void)ws_size;
    const float* cand  = (const float*)d_in[0];
    const float* W_sc0 = (const float*)d_in[1];
    const float* b_sc0 = (const float*)d_in[2];
    const float* W_sc1 = (const float*)d_in[3];
    const float* b_sc1 = (const float*)d_in[4];
    const float* W_fw  = (const float*)d_in[5];
    const float* b_fw  = (const float*)d_in[6];
    const float* W_bw  = (const float*)d_in[7];
    const float* b_bw  = (const float*)d_in[8];
    const float* W_lin = (const float*)d_in[9];
    const float* b_lin = (const float*)d_in[10];

    char* p = (char*)d_ws;
    bf16*  W0cT = (bf16*)p;   p += (size_t)FF * D * 2;
    float* Pa   = (float*)p;  p += (size_t)N * FF * 4;
    float* Pb   = (float*)p;  p += (size_t)N * FF * 4;
    float* Amat = (float*)p;  p += (size_t)R * N * N * 4;
    float* sup  = (float*)p;  p += (size_t)2 * R * N * DH * 4;
    float* y    = (float*)p;  p += (size_t)N * D * 4;
    float* x1   = (float*)p;

    // ---- phase 1: relation scores -> adjacency ----
    k_w0cT<<<(FF * D) / 256, 256, 0, stream>>>(W_sc0, W0cT);
    k_pab <<<(2 * N * FF) / 256, 256, 0, stream>>>(cand, W_sc0, Pa, Pb);
    k_scorer<<<dim3(4, N), 256, 0, stream>>>(cand, Pa, Pb, b_sc0, W_sc1, b_sc1,
                                             W0cT, Amat);

    // ---- phase 2: BiGCN, layer 0 (x = cand -> x1) ----
    k_sup<<<(2 * R * N * DH) / 256, 256, 0, stream>>>(cand, W_fw, W_bw, sup);
    k_gcn<<<(N * D) / 256, 256, 0, stream>>>(Amat, sup, b_fw, b_bw, y);
    k_lin<<<(N * D) / 256, 256, 0, stream>>>(y, W_lin, b_lin, cand, x1);

    // ---- layer 1 (x1 -> d_out) ----
    k_sup<<<(2 * R * N * DH) / 256, 256, 0, stream>>>(
        x1, W_fw + (size_t)R * D * DH, W_bw + (size_t)R * D * DH, sup);
    k_gcn<<<(N * D) / 256, 256, 0, stream>>>(
        Amat, sup, b_fw + (size_t)R * DH, b_bw + (size_t)R * DH, y);
    k_lin<<<(N * D) / 256, 256, 0, stream>>>(
        y, W_lin + (size_t)D * D, b_lin + D, x1, (float*)d_out);
}
